// Reservoir_17600775979433
// MI455X (gfx1250) — matrix operand order repaired and checked by reading, not compiled, not run on hardware
//
#include <hip/hip_runtime.h>
#include <math.h>

#define T_STEPS 256
#define BATCH   64
#define D_IN    768
#define R_RES   2048
#define KC      64                  // K-elements staged per TDM chunk
#define NB      64                  // N columns handled per block
#define NCH_IN  (D_IN / KC)         // 12 chunks of input projection
#define NCH_RES (R_RES / KC)        // 32 chunks of recurrent projection
#define NCHUNK  (NCH_IN + NCH_RES)  // 44

typedef __attribute__((ext_vector_type(16))) __bf16 v16bf;
typedef __attribute__((ext_vector_type(8)))  __bf16 v8bf;
typedef __attribute__((ext_vector_type(8)))  float  v8f;
typedef __attribute__((ext_vector_type(4)))  unsigned int u32x4;
typedef __attribute__((ext_vector_type(4)))  int  i32x4;
typedef __attribute__((ext_vector_type(8)))  int  i32x8;

// ---------------- helpers ----------------

__device__ __forceinline__ void bf16_split(float x, __bf16& hi, __bf16& lo) {
    __bf16 h = (__bf16)x;
    hi = h;
    lo = (__bf16)(x - (float)h);
}

__device__ __forceinline__ unsigned int lds_off(const void* p) {
    // flat address low 32 bits == LDS byte offset (aperture in high bits)
    return (unsigned int)(uintptr_t)p;
}

// TDM 2D tile load: rows x Kc from a row-major [rows][ld] bf16 tensor into
// LDS laid out [row][k] contiguous. D# packed per CDNA5 ISA 8.3/8.4.
__device__ __forceinline__ void tdm_load_2d(unsigned int lds_byte,
                                            const void* gptr,
                                            unsigned int tensor_d0,   // K length (elems)
                                            unsigned int tensor_d1,   // tensor rows
                                            unsigned int tile_d0,     // K chunk (elems)
                                            unsigned int tile_d1,     // rows in tile
                                            unsigned long long stride0) { // row pitch (elems)
    unsigned long long ga = (unsigned long long)(uintptr_t)gptr;
    u32x4 g0;
    g0[0] = 1u;                                          // count=1, user descriptor
    g0[1] = lds_byte;                                    // lds_addr
    g0[2] = (unsigned int)ga;                            // global_addr[31:0]
    g0[3] = ((unsigned int)(ga >> 32) & 0x01FFFFFFu) | 0x80000000u; // addr[56:32] | type=2
    i32x8 g1;
    g1[0] = (int)(1u << 16);                             // data_size=1 -> 2 bytes/elem
    g1[1] = (int)((tensor_d0 & 0xFFFFu) << 16);          // tensor_dim0[15:0]
    g1[2] = (int)(((tensor_d0 >> 16) & 0xFFFFu) | ((tensor_d1 & 0xFFFFu) << 16));
    g1[3] = (int)(((tensor_d1 >> 16) & 0xFFFFu) | ((tile_d0 & 0xFFFFu) << 16));
    g1[4] = (int)(tile_d1 & 0xFFFFu);                    // tile_dim1, tile_dim2=0 (2D)
    g1[5] = (int)(unsigned int)(stride0 & 0xFFFFFFFFull);
    g1[6] = (int)(unsigned int)((stride0 >> 32) & 0xFFFFull);
    g1[7] = 0;
    i32x4 gz = {0, 0, 0, 0};
#if defined(__clang_major__) && (__clang_major__ >= 23)
    i32x8 z8 = {0, 0, 0, 0, 0, 0, 0, 0};
    __builtin_amdgcn_tensor_load_to_lds(g0, g1, gz, gz, z8, 0);
#else
    __builtin_amdgcn_tensor_load_to_lds(g0, g1, gz, gz, 0);
#endif
}

// A-tile 16x32 from LDS slab [row][KC], per ISA A layout:
// lane<16 : M=lane,    halves = K{k0..k0+7, k0+16..k0+23}
// lane>=16: M=lane-16, halves = K{k0+8..k0+15, k0+24..k0+31}
__device__ __forceinline__ v16bf lds_a16x32(const __bf16* base, int row, int k0, int lane) {
    const int khalf = (lane & 16) ? 8 : 0;
    const __bf16* p = base + row * KC + k0 + khalf;
    v8bf c0 = *(const v8bf*)(p);
    v8bf c1 = *(const v8bf*)(p + 16);
    return __builtin_shufflevector(c0, c1, 0,1,2,3,4,5,6,7, 8,9,10,11,12,13,14,15);
}

// B-tile 32x16 from LDS slab [col][KC]:
// lane<16 : N=lane,    halves = K{k0..k0+7, k0+16..k0+23};  lane>=16: N=lane-16, K{k0+8..k0+15, k0+24..k0+31}
__device__ __forceinline__ v16bf lds_b32x16(const __bf16* base, int col, int k0, int lane) {
    const int koff = (lane & 16) ? 8 : 0;
    const __bf16* p = base + col * KC + k0 + koff;
    v8bf c0 = *(const v8bf*)(p);
    v8bf c1 = *(const v8bf*)(p + 16);
    return __builtin_shufflevector(c0, c1, 0,1,2,3,4,5,6,7, 8,9,10,11,12,13,14,15);
}

#define WMMA_BF16(A, B, C) \
    __builtin_amdgcn_wmma_f32_16x16x32_bf16(false, (A), false, (B), (short)0, (C), false, false)

// ---------------- one-time prep kernels ----------------

__global__ void split_f32_to_bf16hl(const float* __restrict__ src,
                                    __bf16* __restrict__ hi,
                                    __bf16* __restrict__ lo, int n) {
    int i = blockIdx.x * blockDim.x + threadIdx.x;
    if (i < n) {
        __bf16 h, l;
        bf16_split(src[i], h, l);
        hi[i] = h;
        lo[i] = l;
    }
}

__global__ void init_state_kernel(const float* __restrict__ x0,
                                  float* __restrict__ x,
                                  __bf16* __restrict__ xh,
                                  __bf16* __restrict__ xl) {
    int i = blockIdx.x * blockDim.x + threadIdx.x;   // exactly B*R threads
    float v = x0[i & (R_RES - 1)];
    x[i] = v;
    __bf16 h, l;
    bf16_split(v, h, l);
    xh[i] = h;
    xl[i] = l;
}

// ---------------- one recurrence step ----------------
// grid: 32 blocks x 256 threads; block owns 64 output columns.
// 8 waves = 4 M-tiles x 2 N-halves; each wave computes 16x32 (two 16x16 tiles).
// TDM double-buffers (A,B) hi/lo K-chunks into LDS; DMA of chunk c+1 overlaps
// WMMA of chunk c; completion via TENSORcnt + workgroup barrier.
__global__ __launch_bounds__(256, 1)
void esn_step(const __bf16* __restrict__ u_hi,  const __bf16* __restrict__ u_lo,     // [B][D] slice t
              const __bf16* __restrict__ win_hi, const __bf16* __restrict__ win_lo,   // [R][D]
              const __bf16* __restrict__ wres_hi, const __bf16* __restrict__ wres_lo, // [R][R]
              const float*  __restrict__ x_old,                                       // [B][R]
              const __bf16* __restrict__ xh_old, const __bf16* __restrict__ xl_old,   // [B][R]
              float* __restrict__ x_new,
              __bf16* __restrict__ xh_new, __bf16* __restrict__ xl_new,
              const float* __restrict__ bias, const int* __restrict__ lengths,
              float* __restrict__ out,                                                // [B][T][R]
              int t) {
    __shared__ __align__(16) __bf16 sA[2][2][BATCH * KC]; // [buf][hi/lo][row*KC+k] 32KB
    __shared__ __align__(16) __bf16 sB[2][2][NB * KC];    // [buf][hi/lo][col*KC+k] 32KB

    const int lane   = threadIdx.x & 31;
    const int wave   = threadIdx.x >> 5;
    const int wave_m = wave & 3;        // 16-row group
    const int wave_n = wave >> 2;       // 0..1 : 32-col half of the block
    const int n0     = blockIdx.x * NB;

    auto issue = [&](int c, int buf) {
        const bool ph0 = (c < NCH_IN);
        const int  k   = (ph0 ? c : c - NCH_IN) * KC;
        const int  ld  = ph0 ? D_IN : R_RES;
        if (wave == 0) {
            const __bf16* a = (ph0 ? u_hi : xh_old) + k;
            tdm_load_2d(lds_off(&sA[buf][0][0]), a, (unsigned)ld, BATCH, KC, BATCH, (unsigned)ld);
        } else if (wave == 1) {
            const __bf16* a = (ph0 ? u_lo : xl_old) + k;
            tdm_load_2d(lds_off(&sA[buf][1][0]), a, (unsigned)ld, BATCH, KC, BATCH, (unsigned)ld);
        } else if (wave == 2) {
            const __bf16* b = (ph0 ? win_hi : wres_hi) + (size_t)n0 * ld + k;
            tdm_load_2d(lds_off(&sB[buf][0][0]), b, (unsigned)ld, R_RES, KC, NB, (unsigned)ld);
        } else if (wave == 3) {
            const __bf16* b = (ph0 ? win_lo : wres_lo) + (size_t)n0 * ld + k;
            tdm_load_2d(lds_off(&sB[buf][1][0]), b, (unsigned)ld, R_RES, KC, NB, (unsigned)ld);
        }
    };

    v8f acc0 = {}, acc1 = {};

    if (wave < 4) issue(0, 0);

    for (int c = 0; c < NCHUNK; ++c) {
        const int buf = c & 1;
        if (wave < 4) __builtin_amdgcn_s_wait_tensorcnt(0);   // chunk c landed
        __syncthreads();                                      // publish buffer
        if (wave < 4 && (c + 1) < NCHUNK) issue(c + 1, buf ^ 1); // overlap next DMA

        const __bf16* Ah = &sA[buf][0][0];
        const __bf16* Al = &sA[buf][1][0];
        const __bf16* Bh = &sB[buf][0][0];
        const __bf16* Bl = &sB[buf][1][0];
        const int arow = wave_m * 16 + (lane & 15);
        const int col0 = wave_n * 32 + (lane & 15);
#pragma unroll
        for (int k32 = 0; k32 < KC; k32 += 32) {
            v16bf ah  = lds_a16x32(Ah, arow, k32, lane);
            v16bf al  = lds_a16x32(Al, arow, k32, lane);
            v16bf bh0 = lds_b32x16(Bh, col0, k32, lane);
            v16bf bl0 = lds_b32x16(Bl, col0, k32, lane);
            acc0 = WMMA_BF16(ah, bh0, acc0);
            acc0 = WMMA_BF16(ah, bl0, acc0);
            acc0 = WMMA_BF16(al, bh0, acc0);
            v16bf bh1 = lds_b32x16(Bh, col0 + 16, k32, lane);
            v16bf bl1 = lds_b32x16(Bl, col0 + 16, k32, lane);
            acc1 = WMMA_BF16(ah, bh1, acc1);
            acc1 = WMMA_BF16(ah, bl1, acc1);
            acc1 = WMMA_BF16(al, bh1, acc1);
        }
        __syncthreads();   // all reads done before buf is overwritten next round
    }

    // ---- epilogue: bias + leaky tanh + length mask + stores ----
    const int mbase = wave_m * 16 + ((lane & 16) ? 8 : 0);
    v8f accs[2] = { acc0, acc1 };
#pragma unroll
    for (int nt = 0; nt < 2; ++nt) {
        const int ncol = n0 + wave_n * 32 + nt * 16 + (lane & 15);
        const float bn = bias[ncol];
#pragma unroll
        for (int i = 0; i < 8; ++i) {
            const int m = mbase + i;
            const size_t idx = (size_t)m * R_RES + ncol;
            float pre = accs[nt][i] + bn;
            float xn  = 0.5f * x_old[idx] + 0.5f * tanhf(pre);  // LEAKING_RATE = 0.5
            xn = (t < lengths[m]) ? xn : 0.0f;
            x_new[idx] = xn;
            __bf16 h, l;
            bf16_split(xn, h, l);
            xh_new[idx] = h;
            xl_new[idx] = l;
            out[((size_t)m * T_STEPS + t) * R_RES + ncol] = xn;
        }
    }
}

// ---------------- host ----------------

extern "C" void kernel_launch(void* const* d_in, const int* in_sizes, int n_in,
                              void* d_out, int out_size, void* d_ws, size_t ws_size,
                              hipStream_t stream) {
    (void)in_sizes; (void)n_in; (void)out_size; (void)ws_size;

    const float* U    = (const float*)d_in[0];   // [T,B,D] f32
    const int*   LEN  = (const int*)  d_in[1];   // [B] i32
    const float* WIN  = (const float*)d_in[2];   // [R,D] f32
    const float* WRES = (const float*)d_in[3];   // [R,R] f32
    const float* BIAS = (const float*)d_in[4];   // [R] f32
    const float* X0   = (const float*)d_in[5];   // [R] f32
    float* OUT = (float*)d_out;                  // [B,T,R] f32

    char*  ws  = (char*)d_ws;
    size_t off = 0;
    auto alloc = [&](size_t bytes) -> void* {
        void* p = ws + off;
        off = (off + bytes + 255) & ~(size_t)255;
        return p;
    };

    const size_t NU  = (size_t)T_STEPS * BATCH * D_IN;
    const size_t NWI = (size_t)R_RES * D_IN;
    const size_t NWR = (size_t)R_RES * R_RES;
    const size_t NX  = (size_t)BATCH * R_RES;

    __bf16* u_hi    = (__bf16*)alloc(NU  * 2);
    __bf16* u_lo    = (__bf16*)alloc(NU  * 2);
    __bf16* win_hi  = (__bf16*)alloc(NWI * 2);
    __bf16* win_lo  = (__bf16*)alloc(NWI * 2);
    __bf16* wres_hi = (__bf16*)alloc(NWR * 2);
    __bf16* wres_lo = (__bf16*)alloc(NWR * 2);
    float*  xbuf[2]  = { (float*)alloc(NX * 4),  (float*)alloc(NX * 4) };
    __bf16* xhbuf[2] = { (__bf16*)alloc(NX * 2), (__bf16*)alloc(NX * 2) };
    __bf16* xlbuf[2] = { (__bf16*)alloc(NX * 2), (__bf16*)alloc(NX * 2) };

    split_f32_to_bf16hl<<<(unsigned)((NU  + 255) / 256), 256, 0, stream>>>(U,    u_hi,    u_lo,    (int)NU);
    split_f32_to_bf16hl<<<(unsigned)((NWI + 255) / 256), 256, 0, stream>>>(WIN,  win_hi,  win_lo,  (int)NWI);
    split_f32_to_bf16hl<<<(unsigned)((NWR + 255) / 256), 256, 0, stream>>>(WRES, wres_hi, wres_lo, (int)NWR);
    init_state_kernel<<<(unsigned)(NX / 256), 256, 0, stream>>>(X0, xbuf[0], xhbuf[0], xlbuf[0]);

    for (int t = 0; t < T_STEPS; ++t) {
        const int r = t & 1;
        const int w = r ^ 1;
        esn_step<<<32, 256, 0, stream>>>(
            u_hi + (size_t)t * BATCH * D_IN,
            u_lo + (size_t)t * BATCH * D_IN,
            win_hi, win_lo, wres_hi, wres_lo,
            xbuf[r], xhbuf[r], xlbuf[r],
            xbuf[w], xhbuf[w], xlbuf[w],
            BIAS, LEN, OUT, t);
    }
}
